// Self_Attention_86612310491177
// MI455X (gfx1250) — matrix operand order repaired and checked by reading, not compiled, not run on hardware
//
#include <hip/hip_runtime.h>
#include <hip/hip_bf16.h>
#include <math.h>

// ---------------------------------------------------------------------------
// MI455X (gfx1250) fused GroupNorm -> QKV -> 8-head attention -> 3x3 conv.
// All GEMM-shaped work via v_wmma_f32_16x16x32_bf16 (wave32 WMMA), with
// 16x64 register blocking per wave so the GEMMs are matrix-core bound, not
// HBM bound (A-fragment reuse x4).
// ---------------------------------------------------------------------------

typedef __attribute__((ext_vector_type(16))) __bf16 v16bf;
typedef __attribute__((ext_vector_type(8)))  __bf16 v8bf;
typedef __attribute__((ext_vector_type(4)))  __bf16 v4bf;
typedef __attribute__((ext_vector_type(8)))  float  v8f;

#define WMMA_BF16(a, b, c) \
  __builtin_amdgcn_wmma_f32_16x16x32_bf16(false, (a), false, (b), (short)0, (c), false, false)

constexpr int BB = 16, HH = 32, WW = 32, CC = 512;
constexpr int S = HH * WW;            // 1024 tokens per batch
constexpr int HEADS = 8, DHEAD = 64;  // HEADS*DHEAD == CC
constexpr int GROUPS = 32;
constexpr float EPS = 1e-5f;

// ---- cross-lane row reductions within 16-lane halves (wave32) --------------
__device__ __forceinline__ float rmax16(float v) {
  v = fmaxf(v, __shfl_xor(v, 1, 16));
  v = fmaxf(v, __shfl_xor(v, 2, 16));
  v = fmaxf(v, __shfl_xor(v, 4, 16));
  v = fmaxf(v, __shfl_xor(v, 8, 16));
  return v;
}
__device__ __forceinline__ float rsum16(float v) {
  v += __shfl_xor(v, 1, 16);
  v += __shfl_xor(v, 2, 16);
  v += __shfl_xor(v, 4, 16);
  v += __shfl_xor(v, 8, 16);
  return v;
}

// ---- assemble a 16x32 A-fragment from two contiguous 8-elem chunks ---------
__device__ __forceinline__ v16bf mk_afrag(v8bf lo, v8bf hi) {
  v16bf a;
#pragma unroll
  for (int e = 0; e < 8; ++e) { a[e] = lo[e]; a[e + 8] = hi[e]; }
  return a;
}

// ===========================================================================
// 1) GroupNorm statistics: one block per (batch, group); float4 streaming.
// ===========================================================================
__global__ __launch_bounds__(256) void gn_stats_kernel(
    const float* __restrict__ x, float* __restrict__ mean, float* __restrict__ rstd) {
  const int bg = blockIdx.x;            // 0..511
  const int b = bg >> 5, g = bg & 31;
  const float* xb = x + (size_t)b * S * CC + g * 16;
  float s = 0.f, ss = 0.f;
  // S*16 elems per group = S*4 float4 chunks
  for (int i = threadIdx.x; i < S * 4; i += 256) {
    const int hw = i >> 2, c4 = (i & 3) << 2;
    const float4 v = *(const float4*)(xb + (size_t)hw * CC + c4);
    s += v.x + v.y + v.z + v.w;
    ss += v.x * v.x + v.y * v.y + v.z * v.z + v.w * v.w;
  }
  __shared__ float red0[256], red1[256];
  red0[threadIdx.x] = s;
  red1[threadIdx.x] = ss;
  __syncthreads();
  for (int off = 128; off > 0; off >>= 1) {
    if ((int)threadIdx.x < off) {
      red0[threadIdx.x] += red0[threadIdx.x + off];
      red1[threadIdx.x] += red1[threadIdx.x + off];
    }
    __syncthreads();
  }
  if (threadIdx.x == 0) {
    const float m = red0[0] * (1.0f / (S * 16.f));
    const float var = red1[0] * (1.0f / (S * 16.f)) - m * m;
    mean[bg] = m;
    rstd[bg] = rsqrtf(var + EPS);
  }
}

// ===========================================================================
// 2) Apply GroupNorm, cast to bf16 token matrix tn[B*S][C]; float4 in, 8B out.
// ===========================================================================
__global__ __launch_bounds__(256) void gn_apply_kernel(
    const float* __restrict__ x, const float* __restrict__ gamma,
    const float* __restrict__ beta, const float* __restrict__ mean,
    const float* __restrict__ rstd, __bf16* __restrict__ tn) {
  const size_t i = (size_t)blockIdx.x * 256 + threadIdx.x;  // (B*S*C)/4 chunks
  const size_t base = i << 2;
  const int c = (int)(base & (CC - 1));
  const int b = (int)(base >> 19);  // base / (S*C)
  const int g = c >> 4;             // 4 consecutive channels stay in one group
  const float m = mean[b * GROUPS + g];
  const float r = rstd[b * GROUPS + g];
  const float4 xv = *(const float4*)(x + base);
  const float4 gv = *(const float4*)(gamma + c);
  const float4 bv = *(const float4*)(beta + c);
  v4bf o;
  o[0] = (__bf16)(((xv.x - m) * r) * gv.x + bv.x);
  o[1] = (__bf16)(((xv.y - m) * r) * gv.y + bv.y);
  o[2] = (__bf16)(((xv.z - m) * r) * gv.z + bv.z);
  o[3] = (__bf16)(((xv.w - m) * r) * gv.w + bv.w);
  *(v4bf*)(tn + base) = o;
}

// ===========================================================================
// 3) Transpose+cast QKV weights: wt[mat][n][k] = w_mat[k][n]  (bf16)
// ===========================================================================
__global__ __launch_bounds__(256) void wqkv_t_kernel(
    const float* __restrict__ wq, const float* __restrict__ wk,
    const float* __restrict__ wv, __bf16* __restrict__ wt) {
  const size_t i = (size_t)blockIdx.x * 256 + threadIdx.x;  // 3*512*512
  const int kk = (int)(i & 511);
  const size_t nn = (i >> 9) & 511;
  const int mat = (int)(i >> 18);
  const float* w = (mat == 0) ? wq : ((mat == 1) ? wk : wv);
  wt[i] = (__bf16)w[(size_t)kk * 512 + nn];
}

// ===========================================================================
// 4) Transpose+cast conv weights: cwt[tap][co][ci] = cw[tap][ci][co] (bf16)
// ===========================================================================
__global__ __launch_bounds__(256) void convw_t_kernel(
    const float* __restrict__ cw, __bf16* __restrict__ cwt) {
  const size_t i = (size_t)blockIdx.x * 256 + threadIdx.x;  // 9*512*512
  const int ci = (int)(i & 511);
  const size_t co = (i >> 9) & 511;
  const size_t tap = i >> 18;
  cwt[i] = (__bf16)cw[(tap * 512 + ci) * 512 + co];
}

// ===========================================================================
// 5) QKV projection GEMM: [16384 x 512] @ [512 x 1536], 16x64 tile per wave
//    (4 accumulators reuse one A fragment). q,k stored [B*S][C]; v stored
//    transposed [B][C][S] for the PV GEMM.
// ===========================================================================
__global__ __launch_bounds__(256) void qkv_gemm_kernel(
    const __bf16* __restrict__ tn, const __bf16* __restrict__ wt,
    __bf16* __restrict__ qb, __bf16* __restrict__ kb, __bf16* __restrict__ vt) {
  const int wave = blockIdx.x * 8 + (threadIdx.x >> 5);  // 24576 waves
  const int lane = threadIdx.x & 31;
  const int M = lane & 15, hl = lane >> 4;
  const int mt = wave / 24;       // 0..1023 (row tiles over B*S)
  const int ng = wave % 24;       // 0..23  (64-wide col groups over 3*512)
  const int mat = ng >> 3;
  const int n0 = (ng & 7) << 6;   // 0..448 within one matrix
  const int bs0 = mt << 4;

  const __bf16* arow = tn + (size_t)(bs0 + M) * CC;
  const __bf16* brow0 = wt + ((size_t)mat * 512 + n0 + M) * CC;

  v8f acc0 = {}, acc1 = {}, acc2 = {}, acc3 = {};
#pragma unroll 2
  for (int k0 = 0; k0 < CC; k0 += 32) {
    __builtin_prefetch(arow + k0 + 64, 0, 1);
    const v16bf a = mk_afrag(*(const v8bf*)(arow + k0 + hl * 16),
                             *(const v8bf*)(arow + k0 + hl * 16 + 8));
    const v16bf b0 = *(const v16bf*)(brow0 + k0 + hl * 16);
    const v16bf b1 = *(const v16bf*)(brow0 + 16 * CC + k0 + hl * 16);
    const v16bf b2 = *(const v16bf*)(brow0 + 32 * CC + k0 + hl * 16);
    const v16bf b3 = *(const v16bf*)(brow0 + 48 * CC + k0 + hl * 16);
    acc0 = WMMA_BF16(a, b0, acc0);
    acc1 = WMMA_BF16(a, b1, acc1);
    acc2 = WMMA_BF16(a, b2, acc2);
    acc3 = WMMA_BF16(a, b3, acc3);
  }

  if (mat < 2) {
    __bf16* ob = (mat == 0) ? qb : kb;
#pragma unroll
    for (int r = 0; r < 8; ++r) {
      const size_t row = (size_t)(bs0 + r + 8 * hl) * CC + n0 + M;
      ob[row]      = (__bf16)acc0[r];
      ob[row + 16] = (__bf16)acc1[r];
      ob[row + 32] = (__bf16)acc2[r];
      ob[row + 48] = (__bf16)acc3[r];
    }
  } else {
    const int b = bs0 >> 10, s0 = bs0 & 1023;
    const v8f* accs[4] = {&acc0, &acc1, &acc2, &acc3};
#pragma unroll
    for (int j = 0; j < 4; ++j) {
      const int ch = n0 + j * 16 + M;
      v8bf ov;
#pragma unroll
      for (int r = 0; r < 8; ++r) ov[r] = (__bf16)(*accs[j])[r];
      *(v8bf*)(vt + ((size_t)b * 512 + ch) * S + s0 + 8 * hl) = ov;
    }
  }
}

// ===========================================================================
// 6) Flash attention: one wave per (batch, head, 16-query tile).
//    Online softmax; P tile re-fragmented through LDS for the PV WMMA.
// ===========================================================================
__global__ __launch_bounds__(256) void attn_kernel(
    const __bf16* __restrict__ qb, const __bf16* __restrict__ kb,
    const __bf16* __restrict__ vt, __bf16* __restrict__ ao) {
  __shared__ __attribute__((aligned(32))) __bf16 lds_p[8][16 * 32];

  const int wib = threadIdx.x >> 5;
  const int lane = threadIdx.x & 31;
  const int M = lane & 15, hl = lane >> 4;
  const int wave = blockIdx.x * 8 + wib;   // 8192 waves
  const int qt = wave & 63;
  const int head = (wave >> 6) & 7;
  const int b = wave >> 9;
  const int q0 = qt << 4;

  // Q A-fragments: dims [0,32) and [32,64)
  v16bf aq0, aq1;
  {
    const __bf16* qrow = qb + (size_t)(b * S + q0 + M) * CC + head * DHEAD;
    aq0 = mk_afrag(*(const v8bf*)(qrow + hl * 16),
                   *(const v8bf*)(qrow + hl * 16 + 8));
    aq1 = mk_afrag(*(const v8bf*)(qrow + 32 + hl * 16),
                   *(const v8bf*)(qrow + 40 + hl * 16));
  }

  float mrow[8], lrow[8];
#pragma unroll
  for (int r = 0; r < 8; ++r) { mrow[r] = -1e30f; lrow[r] = 0.f; }
  v8f acc0 = {}, acc1 = {}, acc2 = {}, acc3 = {};
  const float scale = 0.125f;  // 1/sqrt(64)

  for (int kt2 = 0; kt2 < 32; ++kt2) {
    const int key0 = kt2 * 32;
    // scores for 32 keys = two 16x16 tiles
    v8f sA = {}, sB = {};
    {
      const __bf16* krow = kb + (size_t)(b * S + key0 + M) * CC + head * DHEAD;
      sA = WMMA_BF16(aq0, *(const v16bf*)(krow + hl * 16), sA);
      sA = WMMA_BF16(aq1, *(const v16bf*)(krow + 32 + hl * 16), sA);
    }
    {
      const __bf16* krow = kb + (size_t)(b * S + key0 + 16 + M) * CC + head * DHEAD;
      sB = WMMA_BF16(aq0, *(const v16bf*)(krow + hl * 16), sB);
      sB = WMMA_BF16(aq1, *(const v16bf*)(krow + 32 + hl * 16), sB);
    }
    // online softmax over the combined 16x32 block
#pragma unroll
    for (int r = 0; r < 8; ++r) {
      const float a = sA[r] * scale;
      const float bb2 = sB[r] * scale;
      const float mx = rmax16(fmaxf(a, bb2));
      const float nm = fmaxf(mrow[r], mx);
      const float alpha = __expf(mrow[r] - nm);
      const float pa = __expf(a - nm);
      const float pb = __expf(bb2 - nm);
      lrow[r] = lrow[r] * alpha + rsum16(pa + pb);
      mrow[r] = nm;
      acc0[r] *= alpha; acc1[r] *= alpha; acc2[r] *= alpha; acc3[r] *= alpha;
      lds_p[wib][(r + 8 * hl) * 32 + M] = (__bf16)pa;
      lds_p[wib][(r + 8 * hl) * 32 + 16 + M] = (__bf16)pb;
    }
    // repackage P into an A-fragment (K = 32 keys)
    const __bf16* pr = &lds_p[wib][M * 32];
    const v16bf ap = mk_afrag(*(const v8bf*)(pr + hl * 16),
                              *(const v8bf*)(pr + hl * 16 + 8));
    // PV: V transposed [B][C][S] -> contiguous 32B B-fragments
    const __bf16* vbase =
        vt + ((size_t)b * 512 + head * DHEAD + M) * S + key0 + hl * 16;
    acc0 = WMMA_BF16(ap, *(const v16bf*)(vbase), acc0);
    acc1 = WMMA_BF16(ap, *(const v16bf*)(vbase + 16 * (size_t)S), acc1);
    acc2 = WMMA_BF16(ap, *(const v16bf*)(vbase + 32 * (size_t)S), acc2);
    acc3 = WMMA_BF16(ap, *(const v16bf*)(vbase + 48 * (size_t)S), acc3);
  }

#pragma unroll
  for (int r = 0; r < 8; ++r) {
    const float inv = 1.0f / lrow[r];
    const size_t row = (size_t)(b * S + q0 + r + 8 * hl) * CC + head * DHEAD + M;
    ao[row]      = (__bf16)(acc0[r] * inv);
    ao[row + 16] = (__bf16)(acc1[r] * inv);
    ao[row + 32] = (__bf16)(acc2[r] * inv);
    ao[row + 48] = (__bf16)(acc3[r] * inv);
  }
}

// ===========================================================================
// 7) 3x3 conv as implicit GEMM (K = 9*512), 16x64 tile per wave (A-fragment
//    gather reused across 4 output-channel tiles) + bias + residual, fp32 out.
// ===========================================================================
__global__ __launch_bounds__(256) void conv_kernel(
    const __bf16* __restrict__ ao, const __bf16* __restrict__ cwt,
    const float* __restrict__ cb, const float* __restrict__ xin,
    float* __restrict__ out) {
  const int wave = blockIdx.x * 8 + (threadIdx.x >> 5);  // 8192 waves
  const int lane = threadIdx.x & 31;
  const int M = lane & 15, hl = lane >> 4;
  const int ng = wave & 7, mt = wave >> 3;
  const int n0 = ng << 6;          // 64-wide output-channel group
  const int p0 = mt << 4;
  const int pix = p0 + M;
  const int b = pix >> 10, rem = pix & 1023;
  const int y = rem >> 5, x0 = rem & 31;

  v8f acc0 = {}, acc1 = {}, acc2 = {}, acc3 = {};
  for (int tap = 0; tap < 9; ++tap) {
    const int dy = tap / 3 - 1, dx = tap % 3 - 1;
    const int ny = y + dy, nx = x0 + dx;
    const bool valid = (ny >= 0) & (ny < HH) & (nx >= 0) & (nx < WW);
    const __bf16* arow = ao + (size_t)(b * 1024 + ny * 32 + nx) * CC;
    const __bf16* brow = cwt + ((size_t)tap * 512 + n0 + M) * CC;
#pragma unroll 2
    for (int k0 = 0; k0 < CC; k0 += 32) {
      v8bf lo, hi;
#pragma unroll
      for (int e = 0; e < 8; ++e) { lo[e] = (__bf16)0.0f; hi[e] = (__bf16)0.0f; }
      if (valid) {
        lo = *(const v8bf*)(arow + k0 + hl * 16);
        hi = *(const v8bf*)(arow + k0 + hl * 16 + 8);
      }
      const v16bf a = mk_afrag(lo, hi);
      acc0 = WMMA_BF16(a, *(const v16bf*)(brow + k0 + hl * 16), acc0);
      acc1 = WMMA_BF16(a, *(const v16bf*)(brow + 16 * CC + k0 + hl * 16), acc1);
      acc2 = WMMA_BF16(a, *(const v16bf*)(brow + 32 * CC + k0 + hl * 16), acc2);
      acc3 = WMMA_BF16(a, *(const v16bf*)(brow + 48 * CC + k0 + hl * 16), acc3);
    }
  }
  const int n = n0 + M;
  const float bias0 = cb[n], bias1 = cb[n + 16], bias2 = cb[n + 32], bias3 = cb[n + 48];
#pragma unroll
  for (int r = 0; r < 8; ++r) {
    const size_t op = (size_t)(p0 + r + 8 * hl) * CC + n;
    out[op]      = acc0[r] + bias0 + xin[op];
    out[op + 16] = acc1[r] + bias1 + xin[op + 16];
    out[op + 32] = acc2[r] + bias2 + xin[op + 32];
    out[op + 48] = acc3[r] + bias3 + xin[op + 48];
  }
}

// ===========================================================================
// Host launcher
// ===========================================================================
extern "C" void kernel_launch(void* const* d_in, const int* in_sizes, int n_in,
                              void* d_out, int out_size, void* d_ws, size_t ws_size,
                              hipStream_t stream) {
  (void)in_sizes; (void)n_in; (void)out_size; (void)ws_size;
  const float* x     = (const float*)d_in[0];
  const float* gamma = (const float*)d_in[1];
  const float* beta  = (const float*)d_in[2];
  const float* wq    = (const float*)d_in[3];
  const float* wk    = (const float*)d_in[4];
  const float* wv    = (const float*)d_in[5];
  const float* cw    = (const float*)d_in[6];
  const float* cb    = (const float*)d_in[7];
  float* out = (float*)d_out;

  char* ws = (char*)d_ws;
  size_t off = 0;
  auto alloc = [&](size_t bytes) {
    void* p = ws + off;
    off = (off + bytes + 255) & ~(size_t)255;
    return p;
  };
  const size_t TOK = (size_t)BB * S * CC;  // 8,388,608 elements
  float*  gmean = (float*)alloc(512 * sizeof(float));
  float*  grstd = (float*)alloc(512 * sizeof(float));
  __bf16* tn    = (__bf16*)alloc(TOK * 2);
  __bf16* wt    = (__bf16*)alloc((size_t)3 * 512 * 512 * 2);
  __bf16* cwt   = (__bf16*)alloc((size_t)9 * 512 * 512 * 2);
  __bf16* qb    = (__bf16*)alloc(TOK * 2);
  __bf16* kb    = (__bf16*)alloc(TOK * 2);
  __bf16* vt    = (__bf16*)alloc(TOK * 2);
  __bf16* ao    = (__bf16*)alloc(TOK * 2);

  gn_stats_kernel<<<BB * GROUPS, 256, 0, stream>>>(x, gmean, grstd);
  gn_apply_kernel<<<(int)(TOK / 4 / 256), 256, 0, stream>>>(x, gamma, beta, gmean, grstd, tn);
  wqkv_t_kernel<<<(3 * 512 * 512) / 256, 256, 0, stream>>>(wq, wk, wv, wt);
  convw_t_kernel<<<(9 * 512 * 512) / 256, 256, 0, stream>>>(cw, cwt);
  qkv_gemm_kernel<<<(1024 * 24) / 8, 256, 0, stream>>>(tn, wt, qb, kb, vt);
  attn_kernel<<<8192 / 8, 256, 0, stream>>>(qb, kb, vt, ao);
  conv_kernel<<<8192 / 8, 256, 0, stream>>>(ao, cwt, cb, x, out);
}
